// SCMIL_AttnMoE_58016418235053
// MI455X (gfx1250) — matrix operand order repaired and checked by reading, not compiled, not run on hardware
//
#include <hip/hip_runtime.h>

typedef __attribute__((ext_vector_type(16))) __bf16 v16bf;
typedef __attribute__((ext_vector_type(8)))  float  v8f;

#define N_INST 16384
#define HIDDIM 512

// ---------------------------------------------------------------- WMMA helpers

__device__ __forceinline__ v8f wmma_bf16(v16bf a, v16bf b, v8f c) {
  return __builtin_amdgcn_wmma_f32_16x16x32_bf16(false, a, false, b, (short)0, c,
                                                 false, false);
}

// A fragment (16x32 bf16) from row-major [ld] source (ld multiple of 8 halves).
// Contiguous k order used here: lane m=l&15; lanes>=16 hold K+16; element h holds K+h.
// Per lane: one contiguous 32B run -> ds_load_b128 x2.
__device__ __forceinline__ v16bf load_a_frag(const __bf16* A, int ld, int mbase,
                                             int kbase) {
  const int l = threadIdx.x & 31;
  const int m = mbase + (l & 15);
  const int k8 = (l >> 4) << 4;
  v16bf a;
#pragma unroll
  for (int h = 0; h < 16; ++h) {
    int k = kbase + h + k8;
    a[h] = A[(size_t)m * ld + k];
  }
  return a;
}

// B fragment (32x16 bf16) from a TRANSPOSED staging buffer Bt[n][k] (ld = k
// stride in halves, multiple of 8). Same contiguous k order as A: lane n=l&15; lanes>=16 hold
// K+16. Per lane: one contiguous 32B run -> ds_load_b128 x2.
__device__ __forceinline__ v16bf load_bT_frag(const __bf16* Bt, int ld,
                                              int nbase, int kbase) {
  const int l = threadIdx.x & 31;
  const int n = nbase + (l & 15);
  const int kb = kbase + ((l >> 4) << 4);
  v16bf b;
#pragma unroll
  for (int h = 0; h < 16; ++h) {
    b[h] = Bt[(size_t)n * ld + kb + h];
  }
  return b;
}

__device__ __forceinline__ float half_group_max(float v) {
  v = fmaxf(v, __shfl_xor(v, 1, 32));
  v = fmaxf(v, __shfl_xor(v, 2, 32));
  v = fmaxf(v, __shfl_xor(v, 4, 32));
  v = fmaxf(v, __shfl_xor(v, 8, 32));
  return v;
}
__device__ __forceinline__ float half_group_sum(float v) {
  v += __shfl_xor(v, 1, 32);
  v += __shfl_xor(v, 2, 32);
  v += __shfl_xor(v, 4, 32);
  v += __shfl_xor(v, 8, 32);
  return v;
}
__device__ __forceinline__ float wave_sum(float v) {
  v = half_group_sum(v);
  v += __shfl_xor(v, 16, 32);
  return v;
}

// ------------------------------------------------- generic bf16 WMMA GEMM
// C[M,N] = act(A[M,K] @ B[K,N] + bias). M,N multiples of 128, K arbitrary.
// Block: 256 threads (8 waves), tile 128x128, BK=32, wave grid 2(M)x4(N).
// A staged row-major [128][40], B staged transposed [128 n][40 k]; both give
// b128 fragment loads with bank-staggered rows (stride 20 dwords).
template <bool RELU>
__global__ void __launch_bounds__(256)
gemm_bf16_kernel(const float* __restrict__ A, const float* __restrict__ B,
                 const float* __restrict__ bias, float* __restrict__ C, int M,
                 int K, int N) {
  __shared__ __attribute__((aligned(16))) __bf16 As[128 * 40];
  __shared__ __attribute__((aligned(16))) __bf16 Bt[128 * 40];
  const int m0 = blockIdx.y * 128;
  const int n0 = blockIdx.x * 128;
  const int tid = threadIdx.x;
  const int w = tid >> 5, l = tid & 31;
  const int wm = (w & 1) * 64;
  const int wn = (w >> 1) * 32;
  v8f zero = {};
  v8f acc[4][2];
#pragma unroll
  for (int mt = 0; mt < 4; ++mt) {
    acc[mt][0] = zero;
    acc[mt][1] = zero;
  }
  for (int k0 = 0; k0 < K; k0 += 32) {
    __syncthreads();
#pragma unroll
    for (int i = 0; i < 16; ++i) {
      int idx = tid + i * 256;
      int r = idx >> 5, c = idx & 31;
      int gk = k0 + c;
      As[r * 40 + c] = (__bf16)((gk < K) ? A[(size_t)(m0 + r) * K + gk] : 0.f);
      if (i == 0 && gk + 32 < K)  // prefetch next A k-tile (global_prefetch_b8)
        __builtin_prefetch(&A[(size_t)(m0 + r) * K + gk + 32], 0, 1);
    }
#pragma unroll
    for (int i = 0; i < 16; ++i) {
      int idx = tid + i * 256;
      int r = idx >> 7, c = idx & 127;  // r = k (0..31), c = n (0..127)
      int gk = k0 + r;
      Bt[c * 40 + r] = (__bf16)((gk < K) ? B[(size_t)gk * N + n0 + c] : 0.f);
      if (i == 0 && gk + 32 < K)  // prefetch next B k-tile
        __builtin_prefetch(&B[(size_t)(gk + 32) * N + n0 + c], 0, 1);
    }
    __syncthreads();
    v16bf b0 = load_bT_frag(Bt, 40, wn, 0);
    v16bf b1 = load_bT_frag(Bt, 40, wn + 16, 0);
#pragma unroll
    for (int mt = 0; mt < 4; ++mt) {
      v16bf af = load_a_frag(As, 40, wm + mt * 16, 0);
      acc[mt][0] = wmma_bf16(af, b0, acc[mt][0]);
      acc[mt][1] = wmma_bf16(af, b1, acc[mt][1]);
    }
  }
  const int cn = l & 15, rb = (l >> 4) << 3;
#pragma unroll
  for (int mt = 0; mt < 4; ++mt)
#pragma unroll
    for (int nt = 0; nt < 2; ++nt) {
      int gn = n0 + wn + nt * 16 + cn;
      float bv = bias ? bias[gn] : 0.f;
#pragma unroll
      for (int r = 0; r < 8; ++r) {
        int gm = m0 + wm + mt * 16 + rb + r;
        float v = acc[mt][nt][r] + bv;
        if (RELU) v = fmaxf(v, 0.f);
        C[(size_t)gm * N + gn] = v;
      }
    }
}

// ----------------------------------------------- bilinear resize (E/F proj)
// R[256][16384] = bilinear_resize(W.T [10000,256] -> (256,16384)), half-pixel.
__global__ void __launch_bounds__(256)
resize_kernel(const float* __restrict__ W, float* __restrict__ R) {
  const size_t idx = (size_t)blockIdx.x * 256 + threadIdx.x;
  const int i = (int)(idx >> 14);
  const int j = (int)(idx & 16383);
  float sp = (i + 0.5f) * (10000.f / 256.f) - 0.5f;
  float sq = (j + 0.5f) * (256.f / 16384.f) - 0.5f;
  float pf = floorf(sp), qf = floorf(sq);
  float fp = sp - pf, fq = sq - qf;
  int p0 = (int)pf, q0 = (int)qf;
  int p1 = p0 + 1, q1 = q0 + 1;
  p0 = min(max(p0, 0), 9999);
  p1 = min(max(p1, 0), 9999);
  q0 = min(max(q0, 0), 255);
  q1 = min(max(q1, 0), 255);
  float v00 = W[(size_t)q0 * 10000 + p0];
  float v01 = W[(size_t)q1 * 10000 + p0];
  float v10 = W[(size_t)q0 * 10000 + p1];
  float v11 = W[(size_t)q1 * 10000 + p1];
  R[idx] = (1.f - fp) * ((1.f - fq) * v00 + fq * v01) +
           fp * ((1.f - fq) * v10 + fq * v11);
}

// --------------------------------------------- attention: scores + softmax
// grid (128 row-blocks, 8 heads); P[h][n][k] bf16 output.
// Ks is stored [n][kk] (already B^T layout), row stride 72 halves.
__global__ void __launch_bounds__(256)
attn_scores_kernel(const float* __restrict__ Q, const float* __restrict__ Kc,
                   __bf16* __restrict__ P) {
  __shared__ __attribute__((aligned(16))) __bf16 Qs[128 * 72];
  __shared__ __attribute__((aligned(16))) __bf16 Ks[256 * 72];
  const int m0 = blockIdx.x * 128;
  const int h = blockIdx.y;
  const int tid = threadIdx.x;
#pragma unroll
  for (int i = 0; i < 32; ++i) {
    int idx = tid + i * 256;
    int r = idx >> 6, c = idx & 63;
    Qs[r * 72 + c] = (__bf16)Q[(size_t)(m0 + r) * HIDDIM + h * 64 + c];
  }
#pragma unroll
  for (int i = 0; i < 64; ++i) {
    int idx = tid + i * 256;
    int r = idx >> 6, c = idx & 63;
    Ks[r * 72 + c] = (__bf16)Kc[(size_t)r * HIDDIM + h * 64 + c];
  }
  __syncthreads();
  const int w = tid >> 5, l = tid & 31;
  v8f zero = {};
  v8f s[16];
#pragma unroll
  for (int nt = 0; nt < 16; ++nt) s[nt] = zero;
#pragma unroll
  for (int ks = 0; ks < 2; ++ks) {
    v16bf af = load_a_frag(Qs, 72, w * 16, ks * 32);
#pragma unroll
    for (int nt = 0; nt < 16; ++nt) {
      v16bf bf = load_bT_frag(Ks, 72, nt * 16, ks * 32);
      s[nt] = wmma_bf16(af, bf, s[nt]);
    }
  }
  const int cn = l & 15, rb = (l >> 4) << 3;
  __bf16* Pw = P + ((size_t)h * N_INST + m0 + w * 16) * 256;
#pragma unroll
  for (int r = 0; r < 8; ++r) {
    float mx = -3.0e38f;
#pragma unroll
    for (int nt = 0; nt < 16; ++nt) mx = fmaxf(mx, s[nt][r] * 0.125f);
    mx = half_group_max(mx);
    float sum = 0.f;
#pragma unroll
    for (int nt = 0; nt < 16; ++nt) {
      float e = __expf(s[nt][r] * 0.125f - mx);
      s[nt][r] = e;
      sum += e;
    }
    sum = half_group_sum(sum);
    float inv = 1.f / sum;
#pragma unroll
    for (int nt = 0; nt < 16; ++nt)
      Pw[(size_t)(rb + r) * 256 + nt * 16 + cn] = (__bf16)(s[nt][r] * inv);
  }
}

// ------------------------------------------------ attention: P @ Vc -> AO
// Vc head staged transposed: Vt[d][kk], row stride 264 halves.
__global__ void __launch_bounds__(256)
attn_av_kernel(const __bf16* __restrict__ P, const float* __restrict__ Vc,
               float* __restrict__ AO) {
  __shared__ __attribute__((aligned(16))) __bf16 As[128 * 40];
  __shared__ __attribute__((aligned(16))) __bf16 Vt[64 * 264];
  const int m0 = blockIdx.x * 128;
  const int h = blockIdx.y;
  const int tid = threadIdx.x;
#pragma unroll
  for (int i = 0; i < 64; ++i) {
    int idx = tid + i * 256;
    int r = idx >> 6, c = idx & 63;  // r = kk, c = d
    Vt[c * 264 + r] = (__bf16)Vc[(size_t)r * HIDDIM + h * 64 + c];
  }
  const int w = tid >> 5, l = tid & 31;
  v8f zero = {};
  v8f ao[4] = {zero, zero, zero, zero};
  for (int ks = 0; ks < 8; ++ks) {
    __syncthreads();
#pragma unroll
    for (int i = 0; i < 16; ++i) {
      int idx = tid + i * 256;
      int r = idx >> 5, c = idx & 31;
      As[r * 40 + c] = P[((size_t)h * N_INST + m0 + r) * 256 + ks * 32 + c];
    }
    __syncthreads();
    v16bf af = load_a_frag(As, 40, w * 16, 0);
#pragma unroll
    for (int dt = 0; dt < 4; ++dt) {
      v16bf bf = load_bT_frag(Vt, 264, dt * 16, ks * 32);
      ao[dt] = wmma_bf16(af, bf, ao[dt]);
    }
  }
  const int cn = l & 15, rb = (l >> 4) << 3;
#pragma unroll
  for (int dt = 0; dt < 4; ++dt)
#pragma unroll
    for (int r = 0; r < 8; ++r)
      AO[(size_t)(m0 + w * 16 + rb + r) * HIDDIM + h * 64 + dt * 16 + cn] =
          ao[dt][r];
}

// ------------------------------------- residual + LayerNorm x2 (1 wave/row)
__global__ void __launch_bounds__(256)
ln_kernel(const float* __restrict__ ao, const float* __restrict__ feat,
          const float* __restrict__ g1, const float* __restrict__ b1,
          const float* __restrict__ g2, const float* __restrict__ b2,
          float* __restrict__ att) {
  const int row = blockIdx.x * 8 + (threadIdx.x >> 5);
  const int l = threadIdx.x & 31;
  const float* pa = ao + (size_t)row * HIDDIM;
  const float* pf = feat + (size_t)row * HIDDIM;
  float x[16];
#pragma unroll
  for (int i = 0; i < 16; ++i) x[i] = pa[l + i * 32] + pf[l + i * 32];
#pragma unroll
  for (int pass = 0; pass < 2; ++pass) {
    float s = 0.f;
#pragma unroll
    for (int i = 0; i < 16; ++i) s += x[i];
    float mean = wave_sum(s) * (1.f / HIDDIM);
    float v = 0.f;
#pragma unroll
    for (int i = 0; i < 16; ++i) {
      float d = x[i] - mean;
      v += d * d;
    }
    float var = wave_sum(v) * (1.f / HIDDIM);
    float inv = rsqrtf(var + 1e-5f);
    const float* g = pass ? g2 : g1;
    const float* b = pass ? b2 : b1;
#pragma unroll
    for (int i = 0; i < 16; ++i)
      x[i] = (x[i] - mean) * inv * g[l + i * 32] + b[l + i * 32];
  }
  float* po = att + (size_t)row * HIDDIM;
#pragma unroll
  for (int i = 0; i < 16; ++i) po[l + i * 32] = x[i];
}

// ---------------- MoE: sc[e,n] = tanh(att@W1[e]+b1[e]) . w2[e] + b2[e]
// W1 k-tile staged transposed: Bt[256 n][40 k].
__global__ void __launch_bounds__(256)
moe_kernel(const float* __restrict__ att, const float* __restrict__ w1,
           const float* __restrict__ b1, const float* __restrict__ w2,
           const float* __restrict__ b2, float* __restrict__ sc) {
  __shared__ __attribute__((aligned(16))) __bf16 As[128 * 40];
  __shared__ __attribute__((aligned(16))) __bf16 Bt[256 * 40];
  const int m0 = blockIdx.x * 128;
  const int e = blockIdx.y;
  const float* W1 = w1 + (size_t)e * HIDDIM * 256;
  const int tid = threadIdx.x;
  const int w = tid >> 5, l = tid & 31;
  v8f zero = {};
  v8f acc[16];
#pragma unroll
  for (int nt = 0; nt < 16; ++nt) acc[nt] = zero;
  for (int k0 = 0; k0 < HIDDIM; k0 += 32) {
    __syncthreads();
#pragma unroll
    for (int i = 0; i < 16; ++i) {
      int idx = tid + i * 256;
      int r = idx >> 5, c = idx & 31;
      As[r * 40 + c] = (__bf16)att[(size_t)(m0 + r) * HIDDIM + k0 + c];
    }
#pragma unroll
    for (int i = 0; i < 32; ++i) {
      int idx = tid + i * 256;
      int r = idx >> 8, c = idx & 255;  // r = k (0..31), c = n (0..255)
      Bt[c * 40 + r] = (__bf16)W1[(size_t)(k0 + r) * 256 + c];
    }
    __syncthreads();
    v16bf af = load_a_frag(As, 40, w * 16, 0);
#pragma unroll
    for (int nt = 0; nt < 16; ++nt) {
      v16bf bf = load_bT_frag(Bt, 40, nt * 16, 0);
      acc[nt] = wmma_bf16(af, bf, acc[nt]);
    }
  }
  const int cn = l & 15, rb = (l >> 4) << 3;
#pragma unroll
  for (int r = 0; r < 8; ++r) {
    float t = 0.f;
#pragma unroll
    for (int nt = 0; nt < 16; ++nt) {
      int col = nt * 16 + cn;
      float hm = tanhf(acc[nt][r] + b1[e * 256 + col]);
      t += hm * w2[e * 256 + col];
    }
    t = half_group_sum(t);
    if (cn == 0) sc[(size_t)e * N_INST + m0 + w * 16 + rb + r] = t + b2[e];
  }
}

// ------------------------------------------------------------- small kernels
__global__ void __launch_bounds__(256)
colmean_kernel(const float* __restrict__ att, float* __restrict__ mean) {
  int c = blockIdx.x * 256 + threadIdx.x;
  float s = 0.f;
  for (int n = 0; n < N_INST; ++n) s += att[(size_t)n * HIDDIM + c];
  mean[c] = s * (1.f / N_INST);
}

__global__ void __launch_bounds__(256)
gates_kernel(const float* __restrict__ mean, const float* __restrict__ gw,
             const float* __restrict__ gb, float* __restrict__ gates) {
  __shared__ float t[4];
  if (threadIdx.x < 4) {
    float s = gb[threadIdx.x];
    for (int d = 0; d < HIDDIM; ++d) s += mean[d] * gw[d * 4 + threadIdx.x];
    t[threadIdx.x] = s;
  }
  __syncthreads();
  if (threadIdx.x == 0) {
    float m = fmaxf(fmaxf(t[0], t[1]), fmaxf(t[2], t[3]));
    float e0 = __expf(t[0] - m), e1 = __expf(t[1] - m);
    float e2 = __expf(t[2] - m), e3 = __expf(t[3] - m);
    float ss = e0 + e1 + e2 + e3;
    gates[0] = e0 / ss;
    gates[1] = e1 / ss;
    gates[2] = e2 / ss;
    gates[3] = e3 / ss;
  }
}

__global__ void __launch_bounds__(256)
expstat_kernel(const float* __restrict__ sc, float* __restrict__ stats) {
  __shared__ float red[256];
  const int e = blockIdx.x;
  const int tid = threadIdx.x;
  const float* p = sc + (size_t)e * N_INST;
  float m = -3.0e38f;
  for (int i = tid; i < N_INST; i += 256) m = fmaxf(m, p[i]);
  red[tid] = m;
  __syncthreads();
  for (int s = 128; s > 0; s >>= 1) {
    if (tid < s) red[tid] = fmaxf(red[tid], red[tid + s]);
    __syncthreads();
  }
  m = red[0];
  __syncthreads();
  float su = 0.f;
  for (int i = tid; i < N_INST; i += 256) su += __expf(p[i] - m);
  red[tid] = su;
  __syncthreads();
  for (int s = 128; s > 0; s >>= 1) {
    if (tid < s) red[tid] += red[tid + s];
    __syncthreads();
  }
  if (tid == 0) {
    stats[e * 2] = m;
    stats[e * 2 + 1] = red[0];
  }
}

__global__ void __launch_bounds__(256)
weights_kernel(const float* __restrict__ sc, const float* __restrict__ stats,
               const float* __restrict__ gates, float* __restrict__ wout) {
  const int n = blockIdx.x * 256 + threadIdx.x;
  float s = 0.f;
#pragma unroll
  for (int e = 0; e < 4; ++e)
    s += gates[e] * __expf(sc[(size_t)e * N_INST + n] - stats[e * 2]) /
         stats[e * 2 + 1];
  wout[n] = s;
}

__global__ void __launch_bounds__(256)
bag_kernel(const float* __restrict__ att, const float* __restrict__ wts,
           float* __restrict__ bag) {
  int c = blockIdx.x * 256 + threadIdx.x;
  float s = 0.f;
  for (int n = 0; n < N_INST; ++n) s += wts[n] * att[(size_t)n * HIDDIM + c];
  bag[c] = s;
}

__global__ void __launch_bounds__(256)
cls_kernel(const float* __restrict__ bag, const float* __restrict__ w1,
           const float* __restrict__ b1, const float* __restrict__ w2,
           const float* __restrict__ b2, const float* __restrict__ w3,
           const float* __restrict__ b3, float* __restrict__ disease) {
  __shared__ float c1[256];
  __shared__ float c2[128];
  int t = threadIdx.x;
  {
    float s = b1[t];
    for (int d = 0; d < HIDDIM; ++d) s += bag[d] * w1[d * 256 + t];
    c1[t] = fmaxf(s, 0.f);
  }
  __syncthreads();
  if (t < 128) {
    float s = b2[t];
    for (int d = 0; d < 256; ++d) s += c1[d] * w2[d * 128 + t];
    c2[t] = fmaxf(s, 0.f);
  }
  __syncthreads();
  if (t < 2) {
    float s = b3[t];
    for (int d = 0; d < 128; ++d) s += c2[d] * w3[d * 2 + t];
    disease[t] = s;
  }
}

// ------------------------------------------------------------------ launcher
extern "C" void kernel_launch(void* const* d_in, const int* in_sizes, int n_in,
                              void* d_out, int out_size, void* d_ws,
                              size_t ws_size, hipStream_t stream) {
  (void)in_sizes; (void)n_in; (void)out_size; (void)ws_size;
  const float* x      = (const float*)d_in[0];
  const float* enc_w1 = (const float*)d_in[1];
  const float* enc_b1 = (const float*)d_in[2];
  const float* enc_w2 = (const float*)d_in[3];
  const float* enc_b2 = (const float*)d_in[4];
  const float* enc_w3 = (const float*)d_in[5];
  const float* enc_b3 = (const float*)d_in[6];
  const float* q_w = (const float*)d_in[7];
  const float* q_b = (const float*)d_in[8];
  const float* k_w = (const float*)d_in[9];
  const float* k_b = (const float*)d_in[10];
  const float* v_w = (const float*)d_in[11];
  const float* v_b = (const float*)d_in[12];
  const float* o_w = (const float*)d_in[13];
  const float* o_b = (const float*)d_in[14];
  const float* E_w = (const float*)d_in[15];
  const float* F_w = (const float*)d_in[16];
  const float* ln1_g = (const float*)d_in[17];
  const float* ln1_b = (const float*)d_in[18];
  const float* ln2_g = (const float*)d_in[19];
  const float* ln2_b = (const float*)d_in[20];
  const float* gate_w = (const float*)d_in[21];
  const float* gate_b = (const float*)d_in[22];
  const float* moe_w1 = (const float*)d_in[23];
  const float* moe_b1 = (const float*)d_in[24];
  const float* moe_w2 = (const float*)d_in[25];
  const float* moe_b2 = (const float*)d_in[26];
  const float* cls_w1 = (const float*)d_in[27];
  const float* cls_b1 = (const float*)d_in[28];
  const float* cls_w2 = (const float*)d_in[29];
  const float* cls_b2 = (const float*)d_in[30];
  const float* cls_w3 = (const float*)d_in[31];
  const float* cls_b3 = (const float*)d_in[32];

  float* ws = (float*)d_ws;
  float* out = (float*)d_out;  // [0,512)=bag, [512,514)=disease, [514,+16384)=weights

  const size_t M1 = 1024ull * 1024ull;  // counts are in floats
  float* h1   = ws;              // 16M (dead after enc2)
  float* Qb   = ws;              // 8M  (reuses h1[0:8M])
  float* Kb   = ws + 8 * M1;     // 8M  (reuses h1[8M:16M]; dead after Kc GEMM)
  float* h2   = ws + 16 * M1;    // 8M  (dead after enc3)
  float* Vb   = h2;              // 8M  (reuse; dead after Vc GEMM)
  float* feat = ws + 24 * M1;    // 8M  (live until LN)
  float* Et   = ws + 32 * M1;    // 4M  (dead after Kc GEMM)
  float* Ft   = ws + 36 * M1;    // 4M  (dead after Vc GEMM)
  float* AO   = Et;              // 8M  (reuses Et+Ft region)
  float* Kc   = ws + 40 * M1;    // 256*512
  float* Vc   = Kc + 256 * 512;  // 256*512
  float* attn_out = ws + 41 * M1;  // 8M
  float* att  = Kb;              // 8M  (reuses Kb region)
  float* cmean  = ws + 49 * M1;  // 512
  float* gatesb = cmean + 512;   // 4
  float* statsb = gatesb + 16;   // 8
  float* scb    = statsb + 16;   // 4*16384
  __bf16* Pbuf = (__bf16*)(ws + 50 * M1);  // 8*16384*256 bf16 = 16M float slots

  dim3 blk(256);

  // Linformer E/F bilinear resize (already in GEMM-A layout, i.e. E_m^T)
  resize_kernel<<<16384, blk, 0, stream>>>(E_w, Et);
  resize_kernel<<<16384, blk, 0, stream>>>(F_w, Ft);

  // instance encoder
  gemm_bf16_kernel<true><<<dim3(8, 128), blk, 0, stream>>>(x, enc_w1, enc_b1, h1,
                                                           N_INST, 2000, 1024);
  gemm_bf16_kernel<true><<<dim3(4, 128), blk, 0, stream>>>(h1, enc_w2, enc_b2, h2,
                                                           N_INST, 1024, 512);
  gemm_bf16_kernel<true><<<dim3(4, 128), blk, 0, stream>>>(h2, enc_w3, enc_b3,
                                                           feat, N_INST, 512, 512);

  // Q/K/V projections
  gemm_bf16_kernel<false><<<dim3(4, 128), blk, 0, stream>>>(feat, q_w, q_b, Qb,
                                                            N_INST, 512, 512);
  gemm_bf16_kernel<false><<<dim3(4, 128), blk, 0, stream>>>(feat, k_w, k_b, Kb,
                                                            N_INST, 512, 512);
  gemm_bf16_kernel<false><<<dim3(4, 128), blk, 0, stream>>>(feat, v_w, v_b, Vb,
                                                            N_INST, 512, 512);

  // Kc = Et @ K, Vc = Ft @ V   ([256,16384] x [16384,512])
  gemm_bf16_kernel<false><<<dim3(4, 2), blk, 0, stream>>>(Et, Kb, nullptr, Kc,
                                                          256, N_INST, 512);
  gemm_bf16_kernel<false><<<dim3(4, 2), blk, 0, stream>>>(Ft, Vb, nullptr, Vc,
                                                          256, N_INST, 512);

  // attention: softmax(Q Kc^T / 8) -> P (bf16), then P @ Vc -> AO
  attn_scores_kernel<<<dim3(128, 8), blk, 0, stream>>>(Qb, Kc, Pbuf);
  attn_av_kernel<<<dim3(128, 8), blk, 0, stream>>>(Pbuf, Vc, AO);

  // output projection + residual + double LN
  gemm_bf16_kernel<false><<<dim3(4, 128), blk, 0, stream>>>(AO, o_w, o_b,
                                                            attn_out, N_INST,
                                                            512, 512);
  ln_kernel<<<2048, blk, 0, stream>>>(attn_out, feat, ln1_g, ln1_b, ln2_g, ln2_b,
                                      att);

  // MoE MIL aggregation
  colmean_kernel<<<2, blk, 0, stream>>>(att, cmean);
  gates_kernel<<<1, blk, 0, stream>>>(cmean, gate_w, gate_b, gatesb);
  moe_kernel<<<dim3(128, 4), blk, 0, stream>>>(att, moe_w1, moe_b1, moe_w2,
                                               moe_b2, scb);
  expstat_kernel<<<4, blk, 0, stream>>>(scb, statsb);
  weights_kernel<<<64, blk, 0, stream>>>(scb, statsb, gatesb, out + 514);
  bag_kernel<<<2, blk, 0, stream>>>(att, out + 514, out);

  // classifier
  cls_kernel<<<1, blk, 0, stream>>>(out, cls_w1, cls_b1, cls_w2, cls_b2, cls_w3,
                                    cls_b3, out + 512);
}
